// CCE_Head_29497835389817
// MI455X (gfx1250) — matrix operand order repaired and checked by reading, not compiled, not run on hardware
//
#include <hip/hip_runtime.h>
#include <math.h>

// ---------------- problem constants ----------------
#define B_   8
#define C_   512
#define N_   19
#define HW_  16384
#define KSC  (C_ / 32)    // 16  K-steps over channels
#define KSP  (HW_ / 32)   // 512 K-steps over pixels
#define LN_EPS 1e-5f

typedef __bf16 bf16;
typedef float v8f   __attribute__((ext_vector_type(8)));
typedef bf16  v16bf __attribute__((ext_vector_type(16)));
typedef bf16  v8bf  __attribute__((ext_vector_type(8)));

union Frag { v16bf v; v8bf h[2]; bf16 e[16]; };

__device__ __forceinline__ float sigmoidf(float v) { return 1.0f / (1.0f + __expf(-v)); }

__device__ __forceinline__ v8f wmma_bf16(v16bf a, v16bf b, v8f c) {
  // D = A(16x32 bf16) x B(32x16 bf16) + C(16x16 f32)
  return __builtin_amdgcn_wmma_f32_16x16x32_bf16(
      /*neg_a=*/false, a, /*neg_b=*/false, b,
      /*c_mod=*/(short)0, c, /*reuse_a=*/false, /*reuse_b=*/false);
}

// ---------------- workspace layout (bytes) ----------------
static constexpr size_t OFF_PRE   = 0;                                // probs_pre / probs  [B,N,HW] f32
static constexpr size_t SZ_PRE    = (size_t)B_ * N_ * HW_ * 4;        // 9,961,472
static constexpr size_t OFF_OCR   = OFF_PRE + SZ_PRE;                 // ocr [B,N,C] f32
static constexpr size_t SZ_OCR    = (size_t)B_ * N_ * C_ * 4;         // 311,296
static constexpr size_t OFF_GEFF  = OFF_OCR + SZ_OCR;                 // 1+gate [B,C] f32
static constexpr size_t SZ_GEFF   = (size_t)B_ * C_ * 4;              // 16,384
static constexpr size_t OFF_PACKP = OFF_GEFF + SZ_GEFF;               // proj B-frags bf16 [KSC][6][32][16]
static constexpr size_t SZ_PACKP  = (size_t)KSC * 6 * 32 * 16 * 2;    // 98,304
static constexpr size_t OFF_PACKF = OFF_PACKP + SZ_PACKP;             // final B-frags bf16 [B][KSC][2][32][16]
// total ~10.65 MB

// =====================================================================
// Pack proj weights (map/dist/bnd, each [19,512]) into WMMA B-fragment
// layout: tile t = mat*2 + coltile; per (kstep,tile): 32 lanes x 16 bf16.
// Lane l -> column n = coltile*16 + l%16 ; K = ks*32 + (v/8)*16 + (l<16?0:8) + v%8.
// =====================================================================
__global__ __launch_bounds__(256) void pack_proj_kernel(
    const float* __restrict__ map_w, const float* __restrict__ dist_w,
    const float* __restrict__ bnd_w, bf16* __restrict__ pack)
{
  int idx = blockIdx.x * 256 + threadIdx.x;
  if (idx >= KSC * 6 * 32 * 16) return;
  int v    = idx & 15;
  int lane = (idx >> 4) & 31;
  int tile = (idx >> 9) % 6;
  int ks   = idx / (512 * 6);
  int n    = (tile & 1) * 16 + (lane & 15);
  int mat  = tile >> 1;
  int k    = ks * 32 + ((v >> 3) << 4) + ((lane >> 4) << 3) + (v & 7);
  float w = 0.0f;
  if (n < N_) {
    const float* W = (mat == 0) ? map_w : (mat == 1) ? dist_w : bnd_w;
    w = W[n * C_ + k];
  }
  pack[idx] = (bf16)w;
}

// ---------------------------------------------------------------------
// Staging: load 32 channels x 128 pixels of x (fp32) into a transposed
// bf16 LDS tile. Manually unrolled so all 4 float4 loads issue as one
// clause (MLP) before any DS store; prefetch next K-step's tile.
// Thread tid handles channel rows c = (tid>>5) + {0,8,16,24} at pixel
// quad p4 = (tid&31)*4.
// ---------------------------------------------------------------------
__device__ __forceinline__ void stage_tile(
    bf16 (*tile)[32], const float* __restrict__ xb, int ks, int p0, int tid)
{
  const int c  = tid >> 5;
  const int p4 = (tid & 31) << 2;
  const float* src = &xb[(size_t)(ks * 32 + c) * HW_ + p0 + p4];
  float4 f0 = *(const float4*)(src);
  float4 f1 = *(const float4*)(src + (size_t)8  * HW_);
  float4 f2 = *(const float4*)(src + (size_t)16 * HW_);
  float4 f3 = *(const float4*)(src + (size_t)24 * HW_);
  if (ks + 1 < KSC && (tid & 7) == 0) {                 // one per 128B row chunk
    __builtin_prefetch(src + (size_t)32 * HW_, 0, 1);
    __builtin_prefetch(src + (size_t)40 * HW_, 0, 1);
    __builtin_prefetch(src + (size_t)48 * HW_, 0, 1);
    __builtin_prefetch(src + (size_t)56 * HW_, 0, 1);
  }
  tile[p4 + 0][c     ] = (bf16)f0.x; tile[p4 + 1][c     ] = (bf16)f0.y;
  tile[p4 + 2][c     ] = (bf16)f0.z; tile[p4 + 3][c     ] = (bf16)f0.w;
  tile[p4 + 0][c +  8] = (bf16)f1.x; tile[p4 + 1][c +  8] = (bf16)f1.y;
  tile[p4 + 2][c +  8] = (bf16)f1.z; tile[p4 + 3][c +  8] = (bf16)f1.w;
  tile[p4 + 0][c + 16] = (bf16)f2.x; tile[p4 + 1][c + 16] = (bf16)f2.y;
  tile[p4 + 2][c + 16] = (bf16)f2.z; tile[p4 + 3][c + 16] = (bf16)f2.w;
  tile[p4 + 0][c + 24] = (bf16)f3.x; tile[p4 + 1][c + 24] = (bf16)f3.y;
  tile[p4 + 2][c + 24] = (bf16)f3.z; tile[p4 + 3][c + 24] = (bf16)f3.w;
}

// =====================================================================
// Fused proj GEMM: per block = 128 pixels of one batch; 8 waves x 16 px.
// K loop over C in steps of 32. 6 accumulators: (m,d,bd) x 2 column
// tiles. B-fragment loads are issued BEFORE the staging phase (they do
// not depend on LDS), so their latency is covered by staging + barrier.
// Epilogue applies the sigmoid chain and writes probs_pre[b][n][p].
// =====================================================================
__global__ __launch_bounds__(256) void proj_gemm_kernel(
    const float* __restrict__ x,
    const float* __restrict__ map_b, const float* __restrict__ dist_b,
    const float* __restrict__ bnd_b,
    const bf16* __restrict__ packW, float* __restrict__ pre)
{
  __shared__ __align__(16) bf16 tile[128][32];   // 8 KB
  const int b    = blockIdx.x >> 7;
  const int p0   = (blockIdx.x & 127) << 7;
  const int tid  = threadIdx.x;
  const int lane = tid & 31, wave = tid >> 5;
  const int pl   = lane & 15, half = lane >> 4;

  v8f acc[6] = {};
  const float* xb = x + (size_t)b * C_ * HW_;
  const bf16* bp0 = packW + (size_t)lane * 16;

  for (int ks = 0; ks < KSC; ++ks) {
    // B fragments first: independent of LDS, latency hidden by staging.
    Frag bf[6];
    const bf16* bp = bp0 + (size_t)ks * 6 * 512;
    #pragma unroll
    for (int t = 0; t < 6; ++t) bf[t].v = *(const v16bf*)(bp + (size_t)t * 512);
    __syncthreads();
    stage_tile(tile, xb, ks, p0, tid);
    __syncthreads();
    // A fragment: lane holds pixel pl of this wave's 16-pixel tile.
    Frag a;
    a.h[0] = *(const v8bf*)&tile[wave * 16 + pl][half * 8];
    a.h[1] = *(const v8bf*)&tile[wave * 16 + pl][16 + half * 8];
    #pragma unroll
    for (int t = 0; t < 6; ++t) acc[t] = wmma_bf16(a.v, bf[t].v, acc[t]);
  }

  // Epilogue: sigmoid chain, write pre-softmax logits.
  for (int ct = 0; ct < 2; ++ct) {
    int n = ct * 16 + pl;
    if (n >= N_) continue;
    float mb = map_b[n], db = dist_b[n], bb = bnd_b[n];
    #pragma unroll
    for (int v = 0; v < 8; ++v) {
      int p = p0 + wave * 16 + (half ? v + 8 : v);
      float m  = acc[0 + ct][v] + mb;
      float dd = (acc[2 + ct][v] + db) * sigmoidf(m);
      float bv = (acc[4 + ct][v] + bb) * sigmoidf(dd);
      float d2 = dd + sigmoidf(bv);
      pre[((size_t)b * N_ + n) * HW_ + p] = m + sigmoidf(d2);
    }
  }
}

// =====================================================================
// In-place row softmax over HW per (b,n). Grid = B*N blocks.
// =====================================================================
__global__ __launch_bounds__(256) void softmax_kernel(float* __restrict__ pre)
{
  __shared__ float red[256];
  float* row = pre + (size_t)blockIdx.x * HW_;
  int tid = threadIdx.x;

  float mx = -INFINITY;
  for (int i = tid; i < HW_; i += 256) mx = fmaxf(mx, row[i]);
  red[tid] = mx; __syncthreads();
  for (int s = 128; s; s >>= 1) { if (tid < s) red[tid] = fmaxf(red[tid], red[tid + s]); __syncthreads(); }
  mx = red[0]; __syncthreads();

  float sum = 0.0f;
  for (int i = tid; i < HW_; i += 256) sum += __expf(row[i] - mx);
  red[tid] = sum; __syncthreads();
  for (int s = 128; s; s >>= 1) { if (tid < s) red[tid] += red[tid + s]; __syncthreads(); }
  float inv = 1.0f / red[0]; __syncthreads();

  for (int i = tid; i < HW_; i += 256) row[i] = __expf(row[i] - mx) * inv;
}

// =====================================================================
// ocr[b][n][c] = sum_p probs[b][n][p] * x[b][c][p].
// Block = (b, 16-channel tile); 8 waves split K=16384 pixels; LDS reduce.
// A = probs rows (n), B = x columns (c); float4 loads + bf16 convert.
// =====================================================================
__global__ __launch_bounds__(256) void ocr_kernel(
    const float* __restrict__ probs, const float* __restrict__ x,
    float* __restrict__ ocr)
{
  __shared__ float red[8][2][8][32];            // 16 KB
  const int b    = blockIdx.x >> 5;
  const int c0   = (blockIdx.x & 31) << 4;
  const int tid  = threadIdx.x;
  const int lane = tid & 31, wave = tid >> 5;
  const int nl   = lane & 15, half = lane >> 4;

  v8f acc0 = {}, acc1 = {};
  const float* pb   = probs + (size_t)b * N_ * HW_;
  const float* xrow = x + ((size_t)b * C_ + (c0 + nl)) * HW_ + half * 8;
  const float* prow0 = pb + (size_t)nl * HW_ + half * 8;
  const int n1 = 16 + nl;
  const float* prow1 = (n1 < N_) ? pb + (size_t)n1 * HW_ + half * 8 : prow0;

  for (int ks = wave; ks < KSP; ks += 8) {
    int p0 = ks * 32;
    // B: x[c0+nl][p0 + half*8 .. +7] and [p0 + 16 + half*8 .. +7]  (16 floats, 4x float4)
    Frag bf_;
    {
      const float4* s4 = (const float4*)(xrow + p0);
      float4 f0 = s4[0], f1 = s4[1], f2 = s4[4], f3 = s4[5];
      bf_.e[0]=(bf16)f0.x; bf_.e[1]=(bf16)f0.y; bf_.e[2]=(bf16)f0.z; bf_.e[3]=(bf16)f0.w;
      bf_.e[4]=(bf16)f1.x; bf_.e[5]=(bf16)f1.y; bf_.e[6]=(bf16)f1.z; bf_.e[7]=(bf16)f1.w;
      bf_.e[8]=(bf16)f2.x; bf_.e[9]=(bf16)f2.y; bf_.e[10]=(bf16)f2.z; bf_.e[11]=(bf16)f2.w;
      bf_.e[12]=(bf16)f3.x; bf_.e[13]=(bf16)f3.y; bf_.e[14]=(bf16)f3.z; bf_.e[15]=(bf16)f3.w;
      __builtin_prefetch(xrow + p0 + 8 * 32, 0, 1);   // this wave's next K-step
    }
    // A tile0: rows n = 0..15
    Frag a0;
    {
      const float4* g = (const float4*)(prow0 + p0);
      float4 f0 = g[0], f1 = g[1];
      const float4* h = (const float4*)(prow0 + p0 + 16);
      float4 f2 = h[0], f3 = h[1];
      a0.e[0]=(bf16)f0.x; a0.e[1]=(bf16)f0.y; a0.e[2]=(bf16)f0.z; a0.e[3]=(bf16)f0.w;
      a0.e[4]=(bf16)f1.x; a0.e[5]=(bf16)f1.y; a0.e[6]=(bf16)f1.z; a0.e[7]=(bf16)f1.w;
      a0.e[8]=(bf16)f2.x; a0.e[9]=(bf16)f2.y; a0.e[10]=(bf16)f2.z; a0.e[11]=(bf16)f2.w;
      a0.e[12]=(bf16)f3.x; a0.e[13]=(bf16)f3.y; a0.e[14]=(bf16)f3.z; a0.e[15]=(bf16)f3.w;
    }
    acc0 = wmma_bf16(a0.v, bf_.v, acc0);
    // A tile1: rows n = 16..18, rest zero
    Frag a1;
    if (n1 < N_) {
      const float4* g = (const float4*)(prow1 + p0);
      float4 f0 = g[0], f1 = g[1];
      const float4* h = (const float4*)(prow1 + p0 + 16);
      float4 f2 = h[0], f3 = h[1];
      a1.e[0]=(bf16)f0.x; a1.e[1]=(bf16)f0.y; a1.e[2]=(bf16)f0.z; a1.e[3]=(bf16)f0.w;
      a1.e[4]=(bf16)f1.x; a1.e[5]=(bf16)f1.y; a1.e[6]=(bf16)f1.z; a1.e[7]=(bf16)f1.w;
      a1.e[8]=(bf16)f2.x; a1.e[9]=(bf16)f2.y; a1.e[10]=(bf16)f2.z; a1.e[11]=(bf16)f2.w;
      a1.e[12]=(bf16)f3.x; a1.e[13]=(bf16)f3.y; a1.e[14]=(bf16)f3.z; a1.e[15]=(bf16)f3.w;
    } else {
      #pragma unroll
      for (int j = 0; j < 16; ++j) a1.e[j] = (bf16)0.0f;
    }
    acc1 = wmma_bf16(a1.v, bf_.v, acc1);
  }

  #pragma unroll
  for (int v = 0; v < 8; ++v) { red[wave][0][v][lane] = acc0[v]; red[wave][1][v][lane] = acc1[v]; }
  __syncthreads();

  for (int it = 0; it < 2; ++it) {
    int idx = tid + it * 256;                   // < 512
    int t = idx >> 8, v = (idx >> 5) & 7, l = idx & 31;
    float s = 0.0f;
    for (int w = 0; w < 8; ++w) s += red[w][t][v][l];
    int n = t * 16 + ((l < 16) ? v : v + 8);
    int c = c0 + (l & 15);
    if (n < N_) ocr[((size_t)b * N_ + n) * C_ + c] = s;
  }
}

// =====================================================================
// Head: attention pool over N, 512x512 MLP with LayerNorm+ReLU, sigmoid
// gate. Tiny (4 MFLOP total) -> scalar VALU, one block per batch.
// Writes geff[b][c] = 1 + sigmoid(t2[c]).
// =====================================================================
__global__ __launch_bounds__(256) void head_kernel(
    const float* __restrict__ ocr, const float* __restrict__ mask_w,
    const float* __restrict__ mask_b,
    const float* __restrict__ cm1_w, const float* __restrict__ cm1_b,
    const float* __restrict__ ln_g, const float* __restrict__ ln_b,
    const float* __restrict__ cm2_w, const float* __restrict__ cm2_b,
    float* __restrict__ geff)
{
  __shared__ float s_ocr[N_][C_];               // 38.9 KB
  __shared__ float s_att[N_];
  __shared__ float s_ctx[C_];
  __shared__ float s_t[C_];
  __shared__ float s_red[256];

  const int b = blockIdx.x, tid = threadIdx.x;
  const float* ob = ocr + (size_t)b * N_ * C_;
  for (int i = tid; i < N_ * C_; i += 256) s_ocr[i / C_][i % C_] = ob[i];
  __syncthreads();

  // attention logits: 8 partial threads per row
  if (tid < N_ * 8) {
    int n = tid >> 3, part = tid & 7, cb = part * 64;
    float s = 0.0f;
    for (int c = 0; c < 64; ++c) s += s_ocr[n][cb + c] * mask_w[cb + c];
    s_red[tid] = s;
  }
  __syncthreads();
  if (tid < N_) {
    float s = mask_b[0];
    for (int j = 0; j < 8; ++j) s += s_red[tid * 8 + j];
    s_att[tid] = s;
  }
  __syncthreads();
  if (tid == 0) {
    float mx = s_att[0];
    for (int n = 1; n < N_; ++n) mx = fmaxf(mx, s_att[n]);
    float sum = 0.0f;
    for (int n = 0; n < N_; ++n) { float e = __expf(s_att[n] - mx); s_att[n] = e; sum += e; }
    float inv = 1.0f / sum;
    for (int n = 0; n < N_; ++n) s_att[n] *= inv;
  }
  __syncthreads();

  // ctx[c] = sum_n ocr[n][c] * att[n]
  for (int c = tid; c < C_; c += 256) {
    float s = 0.0f;
    for (int n = 0; n < N_; ++n) s += s_ocr[n][c] * s_att[n];
    s_ctx[c] = s;
  }
  __syncthreads();

  // t = cm1_w @ ctx + cm1_b
  for (int m = tid; m < C_; m += 256) {
    float a = cm1_b[m];
    const float* wr = cm1_w + (size_t)m * C_;
    for (int c = 0; c < C_; ++c) a += s_ctx[c] * wr[c];
    s_t[m] = a;
  }
  __syncthreads();

  // LayerNorm over 512 + ReLU
  s_red[tid] = s_t[tid] + s_t[tid + 256];
  __syncthreads();
  for (int s = 128; s; s >>= 1) { if (tid < s) s_red[tid] += s_red[tid + s]; __syncthreads(); }
  float mu = s_red[0] / (float)C_;
  __syncthreads();
  { float d0 = s_t[tid] - mu, d1 = s_t[tid + 256] - mu; s_red[tid] = d0 * d0 + d1 * d1; }
  __syncthreads();
  for (int s = 128; s; s >>= 1) { if (tid < s) s_red[tid] += s_red[tid + s]; __syncthreads(); }
  float rstd = rsqrtf(s_red[0] / (float)C_ + LN_EPS);
  __syncthreads();
  for (int i = tid; i < C_; i += 256)
    s_t[i] = fmaxf((s_t[i] - mu) * rstd * ln_g[i] + ln_b[i], 0.0f);
  __syncthreads();

  // t2 = cm2_w @ t + cm2_b; geff = 1 + sigmoid(t2)
  for (int c = tid; c < C_; c += 256) {
    float a = cm2_b[c];
    const float* wr = cm2_w + (size_t)c * C_;
    for (int m = 0; m < C_; ++m) a += s_t[m] * wr[m];
    geff[(size_t)b * C_ + c] = 1.0f + sigmoidf(a);
  }
}

// =====================================================================
// Pack per-batch gated final weights: fin_w[n][c] * geff[b][c] -> bf16
// B-fragment layout [b][ks][tile][lane][16].
// =====================================================================
__global__ __launch_bounds__(256) void pack_final_kernel(
    const float* __restrict__ fin_w, const float* __restrict__ geff,
    bf16* __restrict__ pack)
{
  int idx = blockIdx.x * 256 + threadIdx.x;
  if (idx >= B_ * KSC * 2 * 32 * 16) return;
  int v    = idx & 15;
  int lane = (idx >> 4) & 31;
  int tile = (idx >> 9) & 1;
  int ks   = (idx >> 10) & 15;
  int b    = idx >> 14;
  int n = tile * 16 + (lane & 15);
  int k = ks * 32 + ((v >> 3) << 4) + ((lane >> 4) << 3) + (v & 7);
  float w = (n < N_) ? fin_w[n * C_ + k] * geff[b * C_ + k] : 0.0f;
  pack[idx] = (bf16)w;
}

// =====================================================================
// Final conv GEMM: same structure as proj GEMM, with per-batch packed
// gated weights (gate residual folded in). Writes d_out[b][n][p]+fin_b.
// =====================================================================
__global__ __launch_bounds__(256) void final_gemm_kernel(
    const float* __restrict__ x, const bf16* __restrict__ packF,
    const float* __restrict__ fin_b, float* __restrict__ out)
{
  __shared__ __align__(16) bf16 tile[128][32];
  const int b    = blockIdx.x >> 7;
  const int p0   = (blockIdx.x & 127) << 7;
  const int tid  = threadIdx.x;
  const int lane = tid & 31, wave = tid >> 5;
  const int pl   = lane & 15, half = lane >> 4;

  v8f acc[2] = {};
  const float* xb = x + (size_t)b * C_ * HW_;
  const bf16* bp0 = packF + ((size_t)b * KSC * 2) * 512 + (size_t)lane * 16;

  for (int ks = 0; ks < KSC; ++ks) {
    Frag bf[2];
    const bf16* bp = bp0 + (size_t)ks * 2 * 512;
    #pragma unroll
    for (int t = 0; t < 2; ++t) bf[t].v = *(const v16bf*)(bp + (size_t)t * 512);
    __syncthreads();
    stage_tile(tile, xb, ks, p0, tid);
    __syncthreads();
    Frag a;
    a.h[0] = *(const v8bf*)&tile[wave * 16 + pl][half * 8];
    a.h[1] = *(const v8bf*)&tile[wave * 16 + pl][16 + half * 8];
    #pragma unroll
    for (int t = 0; t < 2; ++t) acc[t] = wmma_bf16(a.v, bf[t].v, acc[t]);
  }

  for (int ct = 0; ct < 2; ++ct) {
    int n = ct * 16 + pl;
    if (n >= N_) continue;
    float fb = fin_b[n];
    #pragma unroll
    for (int v = 0; v < 8; ++v) {
      int p = p0 + wave * 16 + (half ? v + 8 : v);
      out[((size_t)b * N_ + n) * HW_ + p] = acc[ct][v] + fb;
    }
  }
}

// =====================================================================
extern "C" void kernel_launch(void* const* d_in, const int* in_sizes, int n_in,
                              void* d_out, int out_size, void* d_ws, size_t ws_size,
                              hipStream_t stream)
{
  const float* x      = (const float*)d_in[0];
  const float* map_w  = (const float*)d_in[1];
  const float* map_b  = (const float*)d_in[2];
  const float* dist_w = (const float*)d_in[3];
  const float* dist_b = (const float*)d_in[4];
  const float* bnd_w  = (const float*)d_in[5];
  const float* bnd_b  = (const float*)d_in[6];
  const float* mask_w = (const float*)d_in[7];
  const float* mask_b = (const float*)d_in[8];
  const float* cm1_w  = (const float*)d_in[9];
  const float* cm1_b  = (const float*)d_in[10];
  const float* ln_g   = (const float*)d_in[11];
  const float* ln_b   = (const float*)d_in[12];
  const float* cm2_w  = (const float*)d_in[13];
  const float* cm2_b  = (const float*)d_in[14];
  const float* fin_w  = (const float*)d_in[15];
  const float* fin_b  = (const float*)d_in[16];

  char* ws = (char*)d_ws;
  float* pre   = (float*)(ws + OFF_PRE);
  float* ocr   = (float*)(ws + OFF_OCR);
  float* geff  = (float*)(ws + OFF_GEFF);
  bf16*  packP = (bf16*)(ws + OFF_PACKP);
  bf16*  packF = (bf16*)(ws + OFF_PACKF);
  float* out   = (float*)d_out;

  // 1. pack proj weights into B-fragment layout (bf16)
  pack_proj_kernel<<<(KSC * 6 * 32 * 16 + 255) / 256, 256, 0, stream>>>(
      map_w, dist_w, bnd_w, packP);
  // 2. fused m/d/bd conv + sigmoid chain -> probs_pre
  proj_gemm_kernel<<<B_ * (HW_ / 128), 256, 0, stream>>>(
      x, map_b, dist_b, bnd_b, packP, pre);
  // 3. softmax over HW (in place)
  softmax_kernel<<<B_ * N_, 256, 0, stream>>>(pre);
  // 4. ocr = probs . x^T
  ocr_kernel<<<B_ * (C_ / 16), 256, 0, stream>>>(pre, x, ocr);
  // 5. attention pool + MLP + LN -> channel gate
  head_kernel<<<B_, 256, 0, stream>>>(
      ocr, mask_w, mask_b, cm1_w, cm1_b, ln_g, ln_b, cm2_w, cm2_b, geff);
  // 6. fold gate into final weights (per batch), pack bf16 B-fragments
  pack_final_kernel<<<(B_ * KSC * 2 * 32 * 16 + 255) / 256, 256, 0, stream>>>(
      fin_w, geff, packF);
  // 7. final conv GEMM -> output
  final_gemm_kernel<<<B_ * (HW_ / 128), 256, 0, stream>>>(
      x, packF, fin_b, out);
}
